// NeuralCDE_4647154614709
// MI455X (gfx1250) — matrix operand order repaired and checked by reading, not compiled, not run on hardware
//
#include <hip/hip_runtime.h>

#define CC   16
#define HH   64
#define HIDN 128
#define LL   128
#define BB   2048

typedef __attribute__((ext_vector_type(16))) __bf16 v16bf;
typedef __attribute__((ext_vector_type(8)))  float  v8f;

union ABf { v16bf v; int4 q[2]; unsigned u[8]; };

// Hardware bf16 convert (v_cvt_pk_bf16_f32) instead of software RNE bit-twiddling.
__device__ __forceinline__ unsigned pack_bf2(float lo, float hi) {
  union { __bf16 b[2]; unsigned u; } r;
  r.b[0] = (__bf16)lo; r.b[1] = (__bf16)hi;
  return r.u;
}

__device__ __forceinline__ float fast_tanh(float x) {
#if __has_builtin(__builtin_amdgcn_tanhf)
  return __builtin_amdgcn_tanhf(x);          // v_tanh_f32 (CDNA5 trans op)
#elif __has_builtin(__builtin_amdgcn_tanh_f32)
  return __builtin_amdgcn_tanh_f32(x);
#else
  return tanhf(x);
#endif
}

// One-time: W1 (128x64) and W2 (1024x128) f32 -> bf16 row-major in workspace.
__global__ void __launch_bounds__(256) prep_weights(const float* __restrict__ W1,
                                                    const float* __restrict__ W2,
                                                    __bf16* __restrict__ w1b,
                                                    __bf16* __restrict__ w2b) {
  int i = blockIdx.x * 256 + threadIdx.x;
  if (i < HIDN * HH)      w1b[i] = (__bf16)W1[i];
  if (i < HH * CC * HIDN) w2b[i] = (__bf16)W2[i];
}

#if __has_builtin(__builtin_amdgcn_global_load_async_to_lds_b128)
typedef int vec4i __attribute__((vector_size(4 * sizeof(int))));  // native clang vector
typedef __attribute__((address_space(1))) vec4i g_vec4i;
typedef __attribute__((address_space(3))) vec4i l_vec4i;
#endif

__global__ void __launch_bounds__(256, 1) cde_kernel(
    const float* __restrict__ coeffs,
    const float* __restrict__ Winit, const float* __restrict__ binit,
    const float* __restrict__ b1,    const float* __restrict__ b2,
    const float* __restrict__ Wout,  const float* __restrict__ bout,
    const __bf16* __restrict__ w1b,
    const __bf16* __restrict__ w2b,
    float* __restrict__ out)
{
  // Weight-stationary in the 320KB CDNA5 LDS: W2 bf16 = 256KB, W1 bf16 = 16KB.
  __shared__ __align__(16) __bf16 sW2[HH * CC * HIDN];  // [n=0..1023][k=0..127]
  __shared__ __align__(16) __bf16 sW1[HIDN * HH];       // [n=0..127][k=0..63]
  __shared__ __align__(16) __bf16 sH [16 * HIDN];       // h (bf16), [m][j]
  __shared__ __align__(16) float sZin  [16 * HH];
  __shared__ __align__(16) float sZbase[16 * HH];
  __shared__ __align__(16) float sAcc  [16 * HH];
  __shared__ __align__(16) float sK    [16 * HH];
  __shared__ __align__(16) float sDX[3][16 * CC];       // d0, dm, d1

  const int tid  = threadIdx.x;
  const int lane = tid & 31;
  const int w    = tid >> 5;                // wave id 0..7
  const int b0   = blockIdx.x * 16;         // batch tile base

  // ---- fill LDS weights (once; reused 127*4 times) ----
#if __has_builtin(__builtin_amdgcn_global_load_async_to_lds_b128)
  {
    g_vec4i* g2 = (g_vec4i*)w2b;
    l_vec4i* l2 = (l_vec4i*)sW2;
    for (int i = tid; i < (HH * CC * HIDN) / 8; i += 256)
      __builtin_amdgcn_global_load_async_to_lds_b128(g2 + i, l2 + i, 0, 0);
    g_vec4i* g1 = (g_vec4i*)w1b;
    l_vec4i* l1 = (l_vec4i*)sW1;
    for (int i = tid; i < (HIDN * HH) / 8; i += 256)
      __builtin_amdgcn_global_load_async_to_lds_b128(g1 + i, l1 + i, 0, 0);
#if __has_builtin(__builtin_amdgcn_s_wait_asynccnt)
    __builtin_amdgcn_s_wait_asynccnt(0);
#else
    asm volatile("s_wait_asynccnt 0" ::: "memory");
#endif
  }
#else
  {
    const int4* g1 = (const int4*)w1b;  int4* l1 = (int4*)sW1;
    for (int i = tid; i < (HIDN * HH) / 8; i += 256) l1[i] = g1[i];
    const int4* g2 = (const int4*)w2b;  int4* l2 = (int4*)sW2;
    for (int i = tid; i < (HH * CC * HIDN) / 8; i += 256) l2[i] = g2[i];
  }
#endif

  // ---- z0 = X0 @ Winit^T + binit ----
  for (int i = tid; i < 16 * HH; i += 256) {
    int m = i >> 6, h = i & 63;
    const float* x0 = coeffs + (size_t)(b0 + m) * (LL - 1) * (4 * CC);
    float s = binit[h];
#pragma unroll
    for (int c = 0; c < CC; ++c) s += x0[c] * Winit[h * CC + c];
    sZbase[i] = s; sZin[i] = s; sAcc[i] = 0.f;
  }
  __syncthreads();

  // ---- per-lane loop invariants (constant across all 127*4 f-evals) ----
  const float b1r = b1[w * 16 + (lane & 15)];
  float b2r[8];
#pragma unroll
  for (int nt = 0; nt < 8; ++nt) b2r[nt] = b2[w * 128 + nt * 16 + (lane & 15)];

  ABf bW1[2];   // W1 B-tiles for this wave's h-columns, kept in VGPRs
  {
    int n = w * 16 + (lane & 15);
#pragma unroll
    for (int kk = 0; kk < 2; ++kk) {
      int kb2 = kk * 32 + ((lane < 16) ? 0 : 8);
      const __bf16* pb = sW1 + n * HH + kb2;
      bW1[kk].q[0] = *(const int4*)(pb);
      bW1[kk].q[1] = *(const int4*)(pb + 16);
    }
  }

  const float cin[4]  = {0.f, 0.5f, 0.5f, 1.f};
  const float wacc[4] = {1.f, 2.f, 2.f, 1.f};
  const int   dsel[4] = {0, 1, 1, 2};

  for (int t = 0; t < LL - 1; ++t) {
    // ---- spline derivs d0 (fr=0), dm (fr=0.5), d1 (fr=0 @ t+1 or fr=1 @ 126) ----
    for (int i = tid; i < 16 * CC; i += 256) {
      int m = i >> 4, c = i & 15;
      const float* seg = coeffs + ((size_t)(b0 + m) * (LL - 1) + t) * (4 * CC);
      float bc = seg[CC + c], tc = seg[2 * CC + c], td = seg[3 * CC + c];
      sDX[0][i] = bc;
      sDX[1][i] = bc + 0.5f * tc + 0.25f * td;
      sDX[2][i] = (t < LL - 2) ? seg[4 * CC + CC + c] : (bc + tc + td);
      if (t + 2 < LL - 1) __builtin_prefetch(seg + 2 * 4 * CC, 0, 0);  // global_prefetch
    }
    __syncthreads();

    for (int s = 0; s < 4; ++s) {
      // ======== f(sZin, dX) -> sK ========
      // matmul1: h = relu(z @ W1^T);  wave w owns h-columns [16w, 16w+16)
      {
        v8f acc = {0.f, 0.f, 0.f, 0.f, 0.f, 0.f, 0.f, 0.f};
#pragma unroll
        for (int kk = 0; kk < 2; ++kk) {
          // A (16x32 bf16) from z f32 in LDS, ISA 16-bit A layout
          ABf a;
          int m  = lane & 15;
          int kb = kk * 32 + ((lane < 16) ? 0 : 8);
          const float* p0 = sZin + m * HH + kb;
#pragma unroll
          for (int j = 0; j < 4; ++j) {
            a.u[j]     = pack_bf2(p0[2 * j],      p0[2 * j + 1]);
            a.u[4 + j] = pack_bf2(p0[16 + 2 * j], p0[16 + 2 * j + 1]);
          }
          acc = __builtin_amdgcn_wmma_f32_16x16x32_bf16(false, a.v, false, bW1[kk].v,
                                                        (short)0, acc, false, false);
        }
        int hi8 = (lane >> 4) << 3;
#pragma unroll
        for (int v = 0; v < 8; ++v) {
          float hv = acc[v] + b1r;
          hv = hv > 0.f ? hv : 0.f;
          sH[(v + hi8) * HIDN + w * 16 + (lane & 15)] = (__bf16)hv;
        }
      }
      __syncthreads();

      // matmul2 + tanh + einsum: wave w owns W2 columns [128w, 128w+128)
      {
        ABf A[4];  // preload all four 16x32 A tiles of h (shared across 8 n-tiles)
        {
          int m   = lane & 15;
          int off = (lane < 16) ? 0 : 8;
#pragma unroll
          for (int kk = 0; kk < 4; ++kk) {
            const __bf16* pa = sH + m * HIDN + kk * 32 + off;
            A[kk].q[0] = *(const int4*)(pa);
            A[kk].q[1] = *(const int4*)(pa + 16);
          }
        }
        const float* dx = sDX[dsel[s]];
        const int hi8 = (lane >> 4) << 3;
        const int kb2 = (lane < 16) ? 0 : 8;
        float dxv[8];                           // hoisted: invariant across nt
#pragma unroll
        for (int v = 0; v < 8; ++v) dxv[v] = dx[(v + hi8) * CC + (lane & 15)];

#pragma unroll
        for (int nt = 0; nt < 8; ++nt) {
          int n0 = w * 128 + nt * 16;
          int n  = n0 + (lane & 15);
          v8f acc = {0.f, 0.f, 0.f, 0.f, 0.f, 0.f, 0.f, 0.f};
#pragma unroll
          for (int kk = 0; kk < 4; ++kk) {
            ABf bm;
            const __bf16* pb = sW2 + n * HIDN + kk * 32 + kb2;
            bm.q[0] = *(const int4*)(pb);
            bm.q[1] = *(const int4*)(pb + 16);
            acc = __builtin_amdgcn_wmma_f32_16x16x32_bf16(false, A[kk].v, false, bm.v,
                                                          (short)0, acc, false, false);
          }
          int hIdx = n0 >> 4;   // each 16-col tile = one h
#pragma unroll
          for (int v = 0; v < 8; ++v) {
            float g = fast_tanh(acc[v] + b2r[nt]);
            float p = g * dxv[v];
#pragma unroll
            for (int off = 1; off < 16; off <<= 1)
              p += __shfl_xor(p, off, 16);          // reduce over c (16 lanes)
            if ((lane & 15) == 0) sK[(v + hi8) * HH + hIdx] = p;
          }
        }
      }
      __syncthreads();

      // RK4 combine
      for (int i = tid; i < 16 * HH; i += 256) {
        float kv = sK[i];
        sAcc[i] += wacc[s] * kv;
        if (s < 3) sZin[i] = sZbase[i] + cin[s + 1] * kv;
      }
      __syncthreads();
    }

    for (int i = tid; i < 16 * HH; i += 256) {
      float zn = sZbase[i] + (1.f / 6.f) * sAcc[i];
      sZbase[i] = zn; sZin[i] = zn; sAcc[i] = 0.f;
    }
    __syncthreads();
  }

  // ---- out = zT @ Wout^T + bout ----
  if (tid < 16) {
    float s = bout[0];
#pragma unroll
    for (int h = 0; h < HH; ++h) s += sZbase[tid * HH + h] * Wout[h];
    out[b0 + tid] = s;
  }
}

extern "C" void kernel_launch(void* const* d_in, const int* in_sizes, int n_in,
                              void* d_out, int out_size, void* d_ws, size_t ws_size,
                              hipStream_t stream) {
  const float* coeffs = (const float*)d_in[0];
  const float* Winit  = (const float*)d_in[1];
  const float* binit  = (const float*)d_in[2];
  const float* W1     = (const float*)d_in[3];
  const float* b1     = (const float*)d_in[4];
  const float* W2     = (const float*)d_in[5];
  const float* b2     = (const float*)d_in[6];
  const float* Wout   = (const float*)d_in[7];
  const float* bout   = (const float*)d_in[8];

  __bf16* w1b = (__bf16*)d_ws;
  __bf16* w2b = w1b + HIDN * HH;

  prep_weights<<<(HH * CC * HIDN + 255) / 256, 256, 0, stream>>>(W1, W2, w1b, w2b);
  cde_kernel<<<BB / 16, 256, 0, stream>>>(coeffs, Winit, binit, b1, b2, Wout, bout,
                                          w1b, w2b, (float*)d_out);
}
